// NESA_5806795784210
// MI455X (gfx1250) — matrix operand order repaired and checked by reading, not compiled, not run on hardware
//
#include <hip/hip_runtime.h>
#include <hip/hip_bf16.h>

#define B_    8
#define N_    1024
#define DIN_  1024
#define H_    8
#define DH_   128
#define DV_   64
#define DENC_ 32
#define QKVC  (2 * DIN_ + H_ * DV_)   // 2560

typedef __attribute__((ext_vector_type(16))) __bf16 bf16x16;
typedef __attribute__((ext_vector_type(8)))  float  f32x8;

union FragU { bf16x16 v; unsigned int d[8]; };

__device__ __forceinline__ unsigned short f2bf(float f) {
  unsigned int u = __float_as_uint(f);
  u += 0x7FFFu + ((u >> 16) & 1u);   // round-to-nearest-even
  return (unsigned short)(u >> 16);
}
__device__ __forceinline__ float bf2f(unsigned short s) {
  return __uint_as_float(((unsigned int)s) << 16);
}
__device__ __forceinline__ f32x8 zero8() {
  f32x8 z = {0.f, 0.f, 0.f, 0.f, 0.f, 0.f, 0.f, 0.f};
  return z;
}

// ---- CDNA5 async global->LDS copy (GV mode, 16B per lane) ------------------
// LDS aperture maps flat addr[31:0] directly to the LDS byte offset (ISA 10.2),
// so truncating the generic pointer gives the VDST LDS address.
__device__ __forceinline__ void async_copy16(void* lds_dst, const void* src) {
  unsigned loff = (unsigned)(size_t)lds_dst;
  asm volatile("global_load_async_to_lds_b128 %0, %1, off"
               :: "v"(loff), "v"(src) : "memory");
}
__device__ __forceinline__ void async_wait0() {
  asm volatile("s_wait_asynccnt 0" ::: "memory");
}

// A-fragment (16x32 bf16, ISA 7.12.2): lane = half*16 + row; VGPR r holds
// K pair at K = (r<4?0:16) + half*8 + 2*(r&3). Source is [row][k] in LDS/mem.
__device__ __forceinline__ bf16x16 load_a_frag(const unsigned short* p, int stride,
                                               int base_row, int kbase, int lane) {
  int m = lane & 15, hf = lane >> 4;
  FragU f;
#pragma unroll
  for (int r = 0; r < 8; ++r) {
    int kk = kbase + ((r < 4) ? 0 : 16) + hf * 8 + 2 * (r & 3);
    f.d[r] = *(const unsigned int*)(p + (size_t)(base_row + m) * stride + kk);
  }
  return f.v;
}
// B-fragment (32x16 bf16): lane = half*16 + col; VGPR r holds K pair at
// K = (r<4?0:16) + half*8 + 2*(r&3). Source stored transposed [col][k].
__device__ __forceinline__ bf16x16 load_b_frag(const unsigned short* p, int stride,
                                               int base_col, int kbase, int lane) {
  int n = lane & 15, hf = lane >> 4;
  FragU f;
#pragma unroll
  for (int r = 0; r < 8; ++r) {
    int kk = kbase + ((r < 4) ? 0 : 16) + hf * 8 + 2 * (r & 3);
    f.d[r] = *(const unsigned int*)(p + (size_t)(base_col + n) * stride + kk);
  }
  return f.v;
}
__device__ __forceinline__ f32x8 wmma_bf16(bf16x16 a, bf16x16 b, f32x8 c) {
  return __builtin_amdgcn_wmma_f32_16x16x32_bf16(false, a, false, b, (short)0, c,
                                                 false, false);
}

#define BM 128
#define BN 128
#define BK 32

// ---------------- Kernel 1: h = x + concat(x, pe) @ W_pos + b_pos -----------
__global__ __launch_bounds__(256) void k_posenc_gemm(
    const float* __restrict__ x, const float* __restrict__ pe,
    const float* __restrict__ W, const float* __restrict__ bias,
    float* __restrict__ hout) {
  __shared__ unsigned short As[BM * BK];
  __shared__ unsigned short Bs[BN * BK];   // stored [n][k]
  const int tid = threadIdx.x, lane = tid & 31, wave = tid >> 5;
  const int wm = wave & 1, wn = wave >> 1;
  const int row0 = blockIdx.y * BM, col0 = blockIdx.x * BN;
  const int K = DIN_ + DENC_;              // 1056
  f32x8 acc[4][2];
#pragma unroll
  for (int i = 0; i < 4; ++i)
#pragma unroll
    for (int j = 0; j < 2; ++j) acc[i][j] = zero8();

  for (int k0 = 0; k0 < K; k0 += BK) {
    __syncthreads();
    // A tile: build concat(x, pe) row-tile, f32 -> packed bf16x2 dwords
    for (int p = tid; p < BM * BK / 2; p += 256) {
      int m = p >> 4;
      int kp = (p & 15) * 2;
      int g = row0 + m, kg = k0 + kp;
      float f0 = (kg < DIN_) ? x[(size_t)g * DIN_ + kg]
                             : pe[(g & (N_ - 1)) * DENC_ + (kg - DIN_)];
      float f1 = (kg + 1 < DIN_) ? x[(size_t)g * DIN_ + kg + 1]
                                 : pe[(g & (N_ - 1)) * DENC_ + (kg + 1 - DIN_)];
      *(unsigned int*)&As[m * BK + kp] =
          (unsigned int)f2bf(f0) | ((unsigned int)f2bf(f1) << 16);
    }
    // B tile: W[k][n] f32 -> LDS [n][k] bf16
    for (int p = tid; p < BN * BK; p += 256) {
      int kkq = p >> 7, n = p & 127;
      Bs[n * BK + kkq] = f2bf(W[(size_t)(k0 + kkq) * DIN_ + (col0 + n)]);
    }
    if (k0 + BK < K)
      __builtin_prefetch(&W[(size_t)(k0 + BK) * DIN_ + col0 + (tid & 127)], 0, 0);
    __syncthreads();
    bf16x16 af[4], bfr[2];
#pragma unroll
    for (int i = 0; i < 4; ++i) af[i] = load_a_frag(As, BK, wm * 64 + i * 16, 0, lane);
#pragma unroll
    for (int j = 0; j < 2; ++j) bfr[j] = load_b_frag(Bs, BK, wn * 32 + j * 16, 0, lane);
#pragma unroll
    for (int i = 0; i < 4; ++i)
#pragma unroll
      for (int j = 0; j < 2; ++j) acc[i][j] = wmma_bf16(af[i], bfr[j], acc[i][j]);
  }
  const int nn = lane & 15, hf = lane >> 4;
#pragma unroll
  for (int i = 0; i < 4; ++i)
#pragma unroll
    for (int j = 0; j < 2; ++j)
#pragma unroll
      for (int r = 0; r < 8; ++r) {
        int g = row0 + wm * 64 + i * 16 + r + hf * 8;
        int c = col0 + wn * 32 + j * 16 + nn;
        hout[(size_t)g * DIN_ + c] = acc[i][j][r] + bias[c] + x[(size_t)g * DIN_ + c];
      }
}

// ---------------- Kernel 2: per-(b,d) mean/var over sequence dim ------------
__global__ __launch_bounds__(256) void k_colstats(const float* __restrict__ h,
                                                  float* __restrict__ mean,
                                                  float* __restrict__ var) {
  int idx = blockIdx.x * 256 + threadIdx.x;   // b*DIN + d, total 8192
  if (idx >= B_ * DIN_) return;
  int b = idx >> 10, d = idx & (DIN_ - 1);
  const float* p = h + (size_t)b * N_ * DIN_ + d;
  float s = 0.f, s2 = 0.f;
  for (int n = 0; n < N_; ++n) {
    float f = p[(size_t)n * DIN_];
    s += f; s2 += f * f;
  }
  float mu = s * (1.0f / N_);
  mean[idx] = mu;
  var[idx] = s2 * (1.0f / N_) - mu * mu;
}

// ---------------- Kernel 3: normalize + convert to bf16 ---------------------
__global__ __launch_bounds__(256) void k_normalize(
    const float* __restrict__ h, const float* __restrict__ mean,
    const float* __restrict__ var, const float* __restrict__ gamma,
    const float* __restrict__ beta, unsigned short* __restrict__ hn) {
  size_t i = (size_t)blockIdx.x * 256 + threadIdx.x;
  if (i >= (size_t)B_ * N_ * DIN_) return;
  int d = (int)(i & (DIN_ - 1));
  int b = (int)(i >> 20);                  // i / (N_*DIN_)
  int sidx = b * DIN_ + d;
  float val = gamma[d] * (h[i] - mean[sidx]) * rsqrtf(var[sidx] + 1e-5f) + beta[d];
  hn[i] = f2bf(val);
}

// ---------------- Kernel 4: QKV GEMM with head-major scatter epilogue -------
__global__ __launch_bounds__(256) void k_qkv_gemm(
    const unsigned short* __restrict__ hn, const float* __restrict__ W,
    const float* __restrict__ bias, unsigned short* __restrict__ qo,
    unsigned short* __restrict__ ko, unsigned short* __restrict__ vo) {
  __shared__ unsigned short As[BM * BK];
  __shared__ unsigned short Bs[BN * BK];
  const int tid = threadIdx.x, lane = tid & 31, wave = tid >> 5;
  const int wm = wave & 1, wn = wave >> 1;
  const int row0 = blockIdx.y * BM, col0 = blockIdx.x * BN;
  f32x8 acc[4][2];
#pragma unroll
  for (int i = 0; i < 4; ++i)
#pragma unroll
    for (int j = 0; j < 2; ++j) acc[i][j] = zero8();

  for (int k0 = 0; k0 < DIN_; k0 += BK) {
    __syncthreads();
    // A tile is raw bf16: async global->LDS DMA, 16B chunks (rows are 64B)
    for (int c = tid; c < BM * 4; c += 256) {
      int m = c >> 2, sub = (c & 3) * 16;
      async_copy16((char*)As + m * (BK * 2) + sub,
                   (const char*)hn + ((size_t)(row0 + m) * DIN_ + k0) * 2 + sub);
    }
    // B tile: f32 weights -> bf16, VALU path
    for (int p = tid; p < BN * BK; p += 256) {
      int kkq = p >> 7, n = p & 127;
      Bs[n * BK + kkq] = f2bf(W[(size_t)(k0 + kkq) * QKVC + (col0 + n)]);
    }
    if (k0 + BK < DIN_)
      __builtin_prefetch(&W[(size_t)(k0 + BK) * QKVC + col0 + (tid & 127)], 0, 0);
    async_wait0();
    __syncthreads();
    bf16x16 af[4], bfr[2];
#pragma unroll
    for (int i = 0; i < 4; ++i) af[i] = load_a_frag(As, BK, wm * 64 + i * 16, 0, lane);
#pragma unroll
    for (int j = 0; j < 2; ++j) bfr[j] = load_b_frag(Bs, BK, wn * 32 + j * 16, 0, lane);
#pragma unroll
    for (int i = 0; i < 4; ++i)
#pragma unroll
      for (int j = 0; j < 2; ++j) acc[i][j] = wmma_bf16(af[i], bfr[j], acc[i][j]);
  }
  const int nn = lane & 15, hf = lane >> 4;
#pragma unroll
  for (int i = 0; i < 4; ++i)
#pragma unroll
    for (int j = 0; j < 2; ++j)
#pragma unroll
      for (int r = 0; r < 8; ++r) {
        int g = row0 + wm * 64 + i * 16 + r + hf * 8;
        int c = col0 + wn * 32 + j * 16 + nn;
        int b = g >> 10, sn = g & (N_ - 1);
        unsigned short val = f2bf(acc[i][j][r] + bias[c]);
        if (c < DIN_) {
          int hh = c >> 7, dh = c & 127;
          qo[(((size_t)b * H_ + hh) * N_ + sn) * DH_ + dh] = val;
        } else if (c < 2 * DIN_) {
          int cc = c - DIN_;
          int hh = cc >> 7, dh = cc & 127;
          ko[(((size_t)b * H_ + hh) * N_ + sn) * DH_ + dh] = val;
        } else {
          int cc = c - 2 * DIN_;
          int hh = cc >> 6, dv = cc & 63;
          vo[(((size_t)b * H_ + hh) * N_ + sn) * DV_ + dv] = val;
        }
      }
}

// ---------------- Kernel 5: sq/sk row reductions ----------------------------
__global__ __launch_bounds__(256) void k_rowsumsq(
    const unsigned short* __restrict__ q, const unsigned short* __restrict__ k,
    float* __restrict__ sq, float* __restrict__ sk) {
  int row = blockIdx.x * 256 + threadIdx.x;   // B*H*N
  if (row >= B_ * H_ * N_) return;
  const unsigned short* qp = q + (size_t)row * DH_;
  const unsigned short* kp = k + (size_t)row * DH_;
  float a = 0.f, c = 0.f;
#pragma unroll 4
  for (int j = 0; j < DH_; ++j) {
    float qf = bf2f(qp[j]); a += qf * qf;
    float kf = bf2f(kp[j]); c += kf * kf;
  }
  sq[row] = a; sk[row] = c;
}

// ---------------- Kernel 6: flash-style distance attention ------------------
// attn = softmax(-(sq + sk - 2 q.k)_+ / scale^2); out = attn @ v
__global__ __launch_bounds__(256) void k_attention(
    const unsigned short* __restrict__ q, const unsigned short* __restrict__ k,
    const unsigned short* __restrict__ v, const float* __restrict__ sq,
    const float* __restrict__ sk, const float* __restrict__ scale_p,
    unsigned short* __restrict__ ao) {
  __shared__ unsigned short Ks[32 * DH_];   // [key][dh]   8KB
  __shared__ unsigned short Vt[DV_ * 32];   // [dv][key]   4KB
  __shared__ float Skc[32];
  __shared__ unsigned short Ps[8][16 * 32]; // per-wave P staging, 8KB
  const int tid = threadIdx.x, lane = tid & 31, wave = tid >> 5;
  const int bh = blockIdx.x;                // b*H + h
  const size_t qbase = (size_t)bh * N_ * DH_;
  const size_t vbase = (size_t)bh * N_ * DV_;
  const int row0 = blockIdx.y * 128 + wave * 16;    // this wave's 16 query rows
  const float s = scale_p[0];
  const float inv_s2 = 1.0f / (s * s);
  const int nn = lane & 15, hf = lane >> 4;

  // Q fragments for the wave's 16 rows (A layout, straight from global)
  bf16x16 qf[4];
  {
    const unsigned short* qrow = q + qbase + (size_t)(row0 + nn) * DH_;
#pragma unroll
    for (int kf = 0; kf < 4; ++kf) {
      FragU f;
#pragma unroll
      for (int r = 0; r < 8; ++r) {
        int kk = kf * 32 + ((r < 4) ? 0 : 16) + hf * 8 + 2 * (r & 3);
        f.d[r] = *(const unsigned int*)&qrow[kk];
      }
      qf[kf] = f.v;
    }
  }
  float sqr[8], mrow[8], lrow[8];
#pragma unroll
  for (int r = 0; r < 8; ++r) {
    sqr[r] = sq[(size_t)bh * N_ + row0 + r + hf * 8];
    mrow[r] = -3.0e38f;
    lrow[r] = 0.0f;
  }
  f32x8 o[4];
#pragma unroll
  for (int i = 0; i < 4; ++i) o[i] = zero8();

  for (int c0 = 0; c0 < N_; c0 += 32) {
    __syncthreads();
    // stage 32 keys x 256B via async global->LDS DMA
    for (int c = tid; c < 32 * 16; c += 256) {
      int key = c >> 4, sub = (c & 15) * 16;
      async_copy16((char*)Ks + key * (DH_ * 2) + sub,
                   (const char*)k + (qbase + (size_t)(c0 + key) * DH_) * 2 + sub);
    }
    // stage V transposed: [dv][key] (VALU path, needs transpose)
    for (int p = tid; p < 32 * DV_; p += 256) {
      int key = p >> 6, dv = p & 63;
      Vt[dv * 32 + key] = v[vbase + (size_t)(c0 + key) * DV_ + dv];
    }
    if (tid < 32) Skc[tid] = sk[(size_t)bh * N_ + c0 + tid];
    async_wait0();
    __syncthreads();

    // S = q @ k^T for two 16-key tiles
    f32x8 st[2];
    st[0] = zero8(); st[1] = zero8();
#pragma unroll
    for (int t = 0; t < 2; ++t)
#pragma unroll
      for (int kf = 0; kf < 4; ++kf) {
        bf16x16 bf = load_b_frag(Ks, DH_, t * 16, kf * 32, lane);
        st[t] = wmma_bf16(qf[kf], bf, st[t]);
      }

    // logits = -max(sq + sk - 2S, 0) / scale^2
    float logit[2][8];
#pragma unroll
    for (int t = 0; t < 2; ++t) {
      float skc = Skc[t * 16 + nn];
#pragma unroll
      for (int r = 0; r < 8; ++r) {
        float d2 = fmaxf(sqr[r] + skc - 2.0f * st[t][r], 0.0f);
        logit[t][r] = -d2 * inv_s2;
      }
    }
    // online softmax; rows live across 16 lanes of each half-wave
#pragma unroll
    for (int r = 0; r < 8; ++r) {
      float mx = fmaxf(logit[0][r], logit[1][r]);
#pragma unroll
      for (int d = 1; d < 16; d <<= 1) mx = fmaxf(mx, __shfl_xor(mx, d, 32));
      float mnew = fmaxf(mrow[r], mx);
      float alpha = __expf(mrow[r] - mnew);
      mrow[r] = mnew;
      lrow[r] *= alpha;
#pragma unroll
      for (int fo = 0; fo < 4; ++fo) o[fo][r] *= alpha;
      float p0 = __expf(logit[0][r] - mnew);
      float p1 = __expf(logit[1][r] - mnew);
      float ps = p0 + p1;
#pragma unroll
      for (int d = 1; d < 16; d <<= 1) ps += __shfl_xor(ps, d, 32);
      lrow[r] += ps;
      int prow = r + 8 * hf;
      Ps[wave][prow * 32 + nn] = f2bf(p0);
      Ps[wave][prow * 32 + 16 + nn] = f2bf(p1);
    }
    // same-wave LDS ops are in-order: safe to read our own staging region
    bf16x16 pf = load_a_frag(Ps[wave], 32, 0, 0, lane);
#pragma unroll
    for (int fo = 0; fo < 4; ++fo) {
      bf16x16 vf = load_b_frag(Vt, 32, fo * 16, 0, lane);
      o[fo] = wmma_bf16(pf, vf, o[fo]);
    }
  }
  // finalize: divide by row sums, write ao[b][n][h*DV + dv] (bf16)
  int b = bh >> 3, hh = bh & 7;
#pragma unroll
  for (int fo = 0; fo < 4; ++fo)
#pragma unroll
    for (int r = 0; r < 8; ++r) {
      int rowg = row0 + r + hf * 8;
      int dv = fo * 16 + nn;
      float val = o[fo][r] / lrow[r];
      ao[((size_t)b * N_ + rowg) * (H_ * DV_) + hh * DV_ + dv] = f2bf(val);
    }
}

// ---------------- Kernel 7: out = x + ao @ W_out + b_out --------------------
__global__ __launch_bounds__(256) void k_out_gemm(
    const unsigned short* __restrict__ ao, const float* __restrict__ W,
    const float* __restrict__ bias, const float* __restrict__ x,
    float* __restrict__ out) {
  __shared__ unsigned short As[BM * BK];
  __shared__ unsigned short Bs[BN * BK];
  const int tid = threadIdx.x, lane = tid & 31, wave = tid >> 5;
  const int wm = wave & 1, wn = wave >> 1;
  const int row0 = blockIdx.y * BM, col0 = blockIdx.x * BN;
  const int K = H_ * DV_;   // 512
  f32x8 acc[4][2];
#pragma unroll
  for (int i = 0; i < 4; ++i)
#pragma unroll
    for (int j = 0; j < 2; ++j) acc[i][j] = zero8();

  for (int k0 = 0; k0 < K; k0 += BK) {
    __syncthreads();
    // A tile raw bf16: async global->LDS DMA
    for (int c = tid; c < BM * 4; c += 256) {
      int m = c >> 2, sub = (c & 3) * 16;
      async_copy16((char*)As + m * (BK * 2) + sub,
                   (const char*)ao + ((size_t)(row0 + m) * K + k0) * 2 + sub);
    }
    for (int p = tid; p < BN * BK; p += 256) {
      int kkq = p >> 7, n = p & 127;
      Bs[n * BK + kkq] = f2bf(W[(size_t)(k0 + kkq) * DIN_ + (col0 + n)]);
    }
    if (k0 + BK < K)
      __builtin_prefetch(&W[(size_t)(k0 + BK) * DIN_ + col0 + (tid & 127)], 0, 0);
    async_wait0();
    __syncthreads();
    bf16x16 af[4], bfr[2];
#pragma unroll
    for (int i = 0; i < 4; ++i) af[i] = load_a_frag(As, BK, wm * 64 + i * 16, 0, lane);
#pragma unroll
    for (int j = 0; j < 2; ++j) bfr[j] = load_b_frag(Bs, BK, wn * 32 + j * 16, 0, lane);
#pragma unroll
    for (int i = 0; i < 4; ++i)
#pragma unroll
      for (int j = 0; j < 2; ++j) acc[i][j] = wmma_bf16(af[i], bfr[j], acc[i][j]);
  }
  const int nn = lane & 15, hf = lane >> 4;
#pragma unroll
  for (int i = 0; i < 4; ++i)
#pragma unroll
    for (int j = 0; j < 2; ++j)
#pragma unroll
      for (int r = 0; r < 8; ++r) {
        int g = row0 + wm * 64 + i * 16 + r + hf * 8;
        int c = col0 + wn * 32 + j * 16 + nn;
        out[(size_t)g * DIN_ + c] = acc[i][j][r] + bias[c] + x[(size_t)g * DIN_ + c];
      }
}

// ---------------- Host-side launcher ----------------------------------------
extern "C" void kernel_launch(void* const* d_in, const int* in_sizes, int n_in,
                              void* d_out, int out_size, void* d_ws, size_t ws_size,
                              hipStream_t stream) {
  (void)in_sizes; (void)n_in; (void)out_size; (void)ws_size;
  const float* x     = (const float*)d_in[0];
  const float* gamma = (const float*)d_in[1];
  const float* beta  = (const float*)d_in[2];
  const float* pe    = (const float*)d_in[3];
  const float* Wpos  = (const float*)d_in[4];
  const float* bpos  = (const float*)d_in[5];
  const float* Wqkv  = (const float*)d_in[6];
  const float* bqkv  = (const float*)d_in[7];
  const float* Wout  = (const float*)d_in[8];
  const float* bout  = (const float*)d_in[9];
  const float* scale = (const float*)d_in[10];

  char* w = (char*)d_ws;
  float* h            = (float*)w;           w += (size_t)B_ * N_ * DIN_ * 4;
  float* mean         = (float*)w;           w += (size_t)B_ * DIN_ * 4;
  float* var          = (float*)w;           w += (size_t)B_ * DIN_ * 4;
  unsigned short* hn  = (unsigned short*)w;  w += (size_t)B_ * N_ * DIN_ * 2;
  unsigned short* qb  = (unsigned short*)w;  w += (size_t)B_ * H_ * N_ * DH_ * 2;
  unsigned short* kb  = (unsigned short*)w;  w += (size_t)B_ * H_ * N_ * DH_ * 2;
  unsigned short* vb  = (unsigned short*)w;  w += (size_t)B_ * H_ * N_ * DV_ * 2;
  float* sqv          = (float*)w;           w += (size_t)B_ * H_ * N_ * 4;
  float* skv          = (float*)w;           w += (size_t)B_ * H_ * N_ * 4;
  unsigned short* ao  = (unsigned short*)w;  w += (size_t)B_ * N_ * H_ * DV_ * 2;

  dim3 blk(256);
  const int Mrows = B_ * N_;   // 8192

  k_posenc_gemm<<<dim3(DIN_ / BN, Mrows / BM), blk, 0, stream>>>(x, pe, Wpos, bpos, h);
  k_colstats<<<dim3((B_ * DIN_ + 255) / 256), blk, 0, stream>>>(h, mean, var);
  k_normalize<<<dim3((unsigned)(((size_t)B_ * N_ * DIN_ + 255) / 256)), blk, 0, stream>>>(
      h, mean, var, gamma, beta, hn);
  k_qkv_gemm<<<dim3(QKVC / BN, Mrows / BM), blk, 0, stream>>>(hn, Wqkv, bqkv, qb, kb, vb);
  k_rowsumsq<<<dim3((B_ * H_ * N_ + 255) / 256), blk, 0, stream>>>(qb, kb, sqv, skv);
  k_attention<<<dim3(B_ * H_, N_ / 128), blk, 0, stream>>>(qb, kb, vb, sqv, skv, scale, ao);
  k_out_gemm<<<dim3(DIN_ / BN, Mrows / BM), blk, 0, stream>>>(ao, Wout, bout, x, (float*)d_out);
}
